// GN_gate_40415642255827
// MI455X (gfx1250) — matrix operand order repaired and checked by reading, not compiled, not run on hardware
//
#include <hip/hip_runtime.h>

typedef __bf16 bf16;
typedef __attribute__((ext_vector_type(16))) __bf16 v16bf;
typedef __attribute__((ext_vector_type(8)))  __bf16 v8bf;
typedef __attribute__((ext_vector_type(8)))  float  v8f;

#define NPX 256        // pixels per tile: 4 image rows x 64 cols
#define CIN 256        // input channels
#define C1  512        // conv1 outputs
#define C2  16         // conv2 outputs
#define XS_STRIDE 264  // padded channel stride for Xs (bank-conflict relief)

// Build a 16-element bf16 fragment from two 16-byte LDS runs
// (callers use contiguous k order, element s = k 16*half + s: VGPR v holds 2 elems).
__device__ __forceinline__ v16bf frag_from_lds(const bf16* p0, const bf16* p1) {
  v8bf lo = *(const v8bf*)p0;
  v8bf hi = *(const v8bf*)p1;
  v16bf r;
#pragma unroll
  for (int i = 0; i < 8; ++i) { r[i] = lo[i]; r[i + 8] = hi[i]; }
  return r;
}

// Fused: relu(W1*X+b1) -> relu(W2*.+b2) -> 4x4 avg pool, one pixel-tile per block.
extern "C" __global__ __launch_bounds__(256)
void moe_fused_conv12_pool(const float* __restrict__ X,
                           const float* __restrict__ W1,
                           const float* __restrict__ B1,
                           const float* __restrict__ W2,
                           const float* __restrict__ B2,
                           float* __restrict__ pooled) {
  extern __shared__ char smem[];
  bf16* Xs   = (bf16*)smem;                 // [256 px][264 ch] bf16 (132 KB)
  bf16* W2s  = Xs + NPX * XS_STRIDE;        // [16][512] bf16     (16 KB)
  bf16* W1s  = W2s + C2 * C1;               // [32][256] bf16     (16 KB)
  float* H2s = (float*)(W1s + 32 * CIN);    // [16][256] f32      (16 KB)

  const int tid   = threadIdx.x;
  const int lane  = tid & 31;
  const int wave  = tid >> 5;               // 8 waves
  const int lhalf = (lane >> 4) & 1;        // wave32 half
  const int lrow  = lane & 15;
  const int b = blockIdx.x >> 4;            // batch
  const int t = blockIdx.x & 15;            // 4-row tile index

  // ---- stage X tile (transpose to [px][ch], f32->bf16) ----
  const float* xb = X + (size_t)b * CIN * 4096 + t * NPX;
  for (int idx = tid; idx < CIN * NPX; idx += 256) {
    int c = idx >> 8;                 // channel
    int p = idx & 255;                // pixel (coalesced global read)
    Xs[p * XS_STRIDE + c] = (bf16)xb[(size_t)c * 4096 + p];
  }
  // ---- stage W2 as bf16 ----
  for (int idx = tid; idx < C2 * C1; idx += 256)
    W2s[idx] = (bf16)W2[idx];

  // H2 accumulators (C/D layout: row M = (lane<16 ? r : r+8)), init with bias b2
  v8f acc2[2];
#pragma unroll
  for (int si = 0; si < 2; ++si)
#pragma unroll
    for (int r = 0; r < 8; ++r)
      acc2[si][r] = B2[r + (lhalf ? 8 : 0)];

  for (int it = 0; it < 16; ++it) {         // o1 chunks of 32
    __syncthreads();                        // W1s safe to overwrite
    const float* w1c = W1 + it * 32 * CIN;
    for (int idx = tid; idx < 32 * CIN; idx += 256)
      W1s[idx] = (bf16)w1c[idx];
    if (it < 15) __builtin_prefetch(W1 + (it + 1) * 32 * CIN + tid * 8, 0, 1);
    __syncthreads();

    // A2 fragment: W2 rows (o2 = lrow), K = this 32-wide o1 chunk
    v16bf a2 = frag_from_lds(&W2s[lrow * C1 + it * 32 + lhalf * 16],
                             &W2s[lrow * C1 + it * 32 + lhalf * 16 + 8]);

#pragma unroll
    for (int si = 0; si < 2; ++si) {
      const int s  = wave + si * 8;         // N-subtile
      const int px = s * 16 + lrow;

      // GEMM1: H1[32 o1 x 16 px] over K=256, C initialized with bias b1
      v8f clo, chi;
#pragma unroll
      for (int r = 0; r < 8; ++r) {
        int m = r + (lhalf ? 8 : 0);
        clo[r] = B1[it * 32 + m];
        chi[r] = B1[it * 32 + 16 + m];
      }
#pragma unroll
      for (int kc = 0; kc < 8; ++kc) {      // K steps of 32
        v16bf a_lo = frag_from_lds(&W1s[lrow * CIN + kc * 32 + lhalf * 16],
                                   &W1s[lrow * CIN + kc * 32 + lhalf * 16 + 8]);
        v16bf a_hi = frag_from_lds(&W1s[(16 + lrow) * CIN + kc * 32 + lhalf * 16],
                                   &W1s[(16 + lrow) * CIN + kc * 32 + lhalf * 16 + 8]);
        v16bf bx   = frag_from_lds(&Xs[px * XS_STRIDE + kc * 32 + lhalf * 16],
                                   &Xs[px * XS_STRIDE + kc * 32 + lhalf * 16 + 8]);
        clo = __builtin_amdgcn_wmma_f32_16x16x32_bf16(false, a_lo, false, bx, (short)0, clo, false, false);
        chi = __builtin_amdgcn_wmma_f32_16x16x32_bf16(false, a_hi, false, bx, (short)0, chi, false, false);
      }

      // relu(H1) -> B fragment for GEMM2 via cross-half lane swap
      // (lane L and L+16 hold complementary M halves of the same column N)
      v16bf b2f;
#pragma unroll
      for (int r = 0; r < 8; ++r) {
        float swl = __shfl_xor(clo[r], 16);
        float swh = __shfl_xor(chi[r], 16);
        float e0 = (lhalf == 0) ? clo[r] : swh;  // K elems 0..7  of this lane
        float e1 = (lhalf == 0) ? swl    : chi[r]; // K elems 8..15
        b2f[r]     = (bf16)fmaxf(e0, 0.0f);
        b2f[r + 8] = (bf16)fmaxf(e1, 0.0f);
      }
      acc2[si] = __builtin_amdgcn_wmma_f32_16x16x32_bf16(false, a2, false, b2f, (short)0, acc2[si], false, false);
    }
  }

  __syncthreads();                          // before writing H2s (own region after W1s)
#pragma unroll
  for (int si = 0; si < 2; ++si) {
    const int px = (wave + si * 8) * 16 + lrow;
#pragma unroll
    for (int r = 0; r < 8; ++r) {
      int m = r + (lhalf ? 8 : 0);
      H2s[m * NPX + px] = fmaxf(acc2[si][r], 0.0f);   // relu(conv2)
    }
  }
  __syncthreads();

  // 4x4 mean pool: thread -> (o2 = tid/16, pool-col j = tid%16); pool-row = t
  {
    int o2 = tid >> 4, j = tid & 15;
    float sum = 0.f;
#pragma unroll
    for (int ro = 0; ro < 4; ++ro)
#pragma unroll
      for (int dc = 0; dc < 4; ++dc)
        sum += H2s[o2 * NPX + ro * 64 + j * 4 + dc];
    pooled[(((b * 16 + o2) * 16 + t) * 16) + j] = sum * (1.0f / 16.0f);
  }
}

// Tiny head: conv3 + gate matmul + top-2 + softmax + CV^2 loss. One block.
extern "C" __global__ __launch_bounds__(256)
void moe_gate_head(const float* __restrict__ pooled,
                   const float* __restrict__ W3,
                   const float* __restrict__ B3,
                   const float* __restrict__ WG,
                   float* __restrict__ out) {
  __shared__ float feats[4096];
  __shared__ float part[256];
  __shared__ float logitsS[64];
  __shared__ float gatesS[64];
  const int tid = threadIdx.x;

  for (int b = 0; b < 16; ++b) {
    // conv3 (16->16 channels over 256 positions)
    for (int idx = tid; idx < 4096; idx += 256) {
      int o = idx >> 8, yx = idx & 255;
      float acc = B3[o];
#pragma unroll
      for (int c = 0; c < 16; ++c)
        acc += W3[o * 16 + c] * pooled[((b * 16 + c) << 8) + yx];
      feats[idx] = acc;
    }
    __syncthreads();
    // logits[b][e] = feats . WG[:,e]  (64 partial chunks per expert)
    {
      int e = tid & 3, chunk = tid >> 2;
      float acc = 0.f;
      for (int f = chunk * 64; f < chunk * 64 + 64; ++f)
        acc += feats[f] * WG[f * 4 + e];
      part[tid] = acc;
    }
    __syncthreads();
    if (tid < 4) {
      float acc = 0.f;
      for (int c = 0; c < 64; ++c) acc += part[c * 4 + tid];
      logitsS[b * 4 + tid] = acc;
    }
    __syncthreads();
  }

  if (tid < 16) {                           // per-batch top-2 + softmax
    const int b = tid;
    float v[4];
#pragma unroll
    for (int e = 0; e < 4; ++e) v[e] = logitsS[b * 4 + e];
    int i0 = 0;
#pragma unroll
    for (int e = 1; e < 4; ++e) if (v[e] > v[i0]) i0 = e;   // ties: lowest index
    int i1 = (i0 == 0) ? 1 : 0;
#pragma unroll
    for (int e = 0; e < 4; ++e) if (e != i0 && v[e] > v[i1]) i1 = e;
    float p0 = 1.0f / (1.0f + __expf(v[i1] - v[i0]));
    float p1 = 1.0f - p0;
    out[b * 2 + 0] = p0;                    // expert_gates [16,2]
    out[b * 2 + 1] = p1;
    out[32 + b * 2 + 0] = (float)i0;        // expert_indices [16,2]
    out[32 + b * 2 + 1] = (float)i1;
#pragma unroll
    for (int e = 0; e < 4; ++e) gatesS[b * 4 + e] = 0.f;
    gatesS[b * 4 + i0] = p0;
    gatesS[b * 4 + i1] = p1;
  }
  __syncthreads();
  if (tid == 0) {
    float loss = 0.f;
    for (int pass = 0; pass < 2; ++pass) {  // 0: importance, 1: load
      float val[4], s = 0.f;
      for (int e = 0; e < 4; ++e) {
        float a = 0.f;
        for (int bb = 0; bb < 16; ++bb) {
          float g = gatesS[bb * 4 + e];
          a += (pass == 0) ? g : (g > 0.f ? 1.f : 0.f);
        }
        val[e] = a; s += a;
      }
      float mean = s * 0.25f, var = 0.f;
      for (int e = 0; e < 4; ++e) { float d = val[e] - mean; var += d * d; }
      var *= (1.0f / 3.0f);                 // ddof=1
      loss += var / (mean * mean + 1e-10f);
    }
    out[64] = loss;
  }
}

extern "C" void kernel_launch(void* const* d_in, const int* in_sizes, int n_in,
                              void* d_out, int out_size, void* d_ws, size_t ws_size,
                              hipStream_t stream) {
  const float* x  = (const float*)d_in[0];
  const float* w1 = (const float*)d_in[1];
  const float* b1 = (const float*)d_in[2];
  const float* w2 = (const float*)d_in[3];
  const float* b2 = (const float*)d_in[4];
  const float* w3 = (const float*)d_in[5];
  const float* b3 = (const float*)d_in[6];
  const float* wg = (const float*)d_in[7];
  float* pooled = (float*)d_ws;             // [16][16][16][16] f32 = 256 KB scratch

  size_t lds = (size_t)(NPX * XS_STRIDE + C2 * C1 + 32 * CIN) * sizeof(bf16)
             + (size_t)(C2 * NPX) * sizeof(float);   // 180 KB (< 320 KB/WGP)
  moe_fused_conv12_pool<<<dim3(16 * 16), dim3(256), lds, stream>>>(x, w1, b1, w2, b2, pooled);
  moe_gate_head<<<dim3(1), dim3(256), 0, stream>>>(pooled, w3, b3, wg, (float*)d_out);
}
